// SelfAttention_82308753260995
// MI455X (gfx1250) — matrix operand order repaired and checked by reading, not compiled, not run on hardware
//
#include <hip/hip_runtime.h>
#include <hip/hip_bf16.h>

// ---- geometry -------------------------------------------------------------
#define BB 8
#define LL 768
#define DD 256
#define HH 8
#define CC 32
#define NTOK (BB * LL)          // 6144
#define OUT0 (NTOK * DD)        // 1,572,864 floats (out)
#define NEG_BIG (-1.0e9f)
#define QSCALE 0.17677669529663687f   // 32^-0.5
#define LPAD 772                // LDS row stride (f32), keeps float4 alignment

typedef unsigned short u16;
typedef __attribute__((ext_vector_type(8)))  unsigned short ushort8;
typedef __attribute__((ext_vector_type(16))) unsigned short ushort16;
typedef __attribute__((ext_vector_type(16))) __bf16 v16bf;
typedef __attribute__((ext_vector_type(8)))  float  v8f;

// fp32 -> bf16, round-to-nearest-even
__device__ __forceinline__ u16 f2bf(float f) {
  unsigned int x = __builtin_bit_cast(unsigned int, f);
  unsigned int r = x + 0x7FFFu + ((x >> 16) & 1u);
  return (u16)(r >> 16);
}

// A-fragment (16x32 bf16): lane holds row (lane&15); element e -> K =
// (lane>>4)*8 + (e>>3)*16 + (e&7). base points at (row, kbase); chunks at
// +0 and +16 elements.
__device__ __forceinline__ v16bf load_frag_a(const u16* base) {
  ushort8 lo = *(const ushort8*)(base);
  ushort8 hi = *(const ushort8*)(base + 16);
  ushort16 u = __builtin_shufflevector(lo, hi, 0,1,2,3,4,5,6,7,8,9,10,11,12,13,14,15);
  return __builtin_bit_cast(v16bf, u);
}

// B-fragment (32x16 bf16): lane holds column (lane&15); element e -> K =
// (lane>>4)*8 + (e>>3)*16 + (e&7). base points at (col, kbase); chunks at
// +0 and +16 elements.
__device__ __forceinline__ v16bf load_frag_b(const u16* base) {
  ushort8 lo = *(const ushort8*)(base);
  ushort8 hi = *(const ushort8*)(base + 16);
  ushort16 u = __builtin_shufflevector(lo, hi, 0,1,2,3,4,5,6,7,8,9,10,11,12,13,14,15);
  return __builtin_bit_cast(v16bf, u);
}

// ---- fp32 -> bf16 conversion ----------------------------------------------
__global__ void cvt_bf16_kernel(const float* __restrict__ src, u16* __restrict__ dst, int n) {
  int i = blockIdx.x * 256 + threadIdx.x;
  if (i < n) dst[i] = f2bf(src[i]);
}

// ---- generic WMMA GEMM: out = A[M,256] * W[256,256]^T  (W is [out,in]) ----
// 8 waves / block; wave computes 16(M) x 64(N). Modes select the epilogue:
//   0: store bf16 [M,256]                 (K projection)
//   1: store bf16 * QSCALE                (Q projection, pre-scaled)
//   2: store bf16 transposed [B,H,C,L]    (V projection)
//   3: sigmoid(x + bg) -> f32 [M,256]     (gate)
//   4: store f32 [M,256]                  (final y @ Wo^T -> d_out)
__global__ __launch_bounds__(256)
void gemm_bf16_kernel(const u16* __restrict__ A, const u16* __restrict__ W, int mode,
                      const float* __restrict__ bgv, float* __restrict__ outF,
                      u16* __restrict__ outB) {
  const int lane = threadIdx.x & 31;
  const int wv   = threadIdx.x >> 5;
  const int m0   = blockIdx.x * 128 + wv * 16;
  const int n0   = blockIdx.y * 64;
  const int row  = lane & 15;
  const int half = lane >> 4;
  const int kbase = half * 8;
  const int crow  = half * 8;

  v8f acc[4];
  #pragma unroll
  for (int t = 0; t < 4; ++t) acc[t] = (v8f)(0.0f);

  for (int k0 = 0; k0 < 256; k0 += 32) {
    const u16* ap = A + (size_t)(m0 + row) * DD + k0 + kbase;
    v16bf af = load_frag_a(ap);
    #pragma unroll
    for (int t = 0; t < 4; ++t) {
      const u16* bp = W + (size_t)(n0 + t * 16 + row) * DD + k0 + crow;
      v16bf bf = load_frag_b(bp);
      acc[t] = __builtin_amdgcn_wmma_f32_16x16x32_bf16(
          false, af, false, bf, (short)0, acc[t], false, false);
    }
  }

  #pragma unroll
  for (int t = 0; t < 4; ++t) {
    #pragma unroll
    for (int j = 0; j < 8; ++j) {
      int M = m0 + half * 8 + j;
      int N = n0 + t * 16 + row;
      float v = acc[t][j];
      if (mode == 0) {
        outB[(size_t)M * DD + N] = f2bf(v);
      } else if (mode == 1) {
        outB[(size_t)M * DD + N] = f2bf(v * QSCALE);
      } else if (mode == 2) {
        int bb = M / LL, l = M % LL, hh = N >> 5, c = N & 31;
        outB[((size_t)((bb * HH + hh) * CC + c)) * LL + l] = f2bf(v);
      } else if (mode == 3) {
        float g = 1.0f / (1.0f + __expf(-(v + bgv[N])));
        outF[(size_t)M * DD + N] = g;
      } else {
        outF[(size_t)M * DD + N] = v;
      }
    }
  }
}

// ---- attention: 4 waves cooperate on 16 q-rows of one (b,h) ----------------
// Shared 48KB score tile, split-K everywhere -> ~12KB LDS per wave.
__global__ __launch_bounds__(128)
void attn_kernel(const u16* __restrict__ Qb,   // [NTOK, D] bf16, pre-scaled
                 const u16* __restrict__ Kb,   // [NTOK, D] bf16
                 const u16* __restrict__ Vt,   // [B,H,C,L] bf16 (transposed)
                 const float* __restrict__ G,  // [NTOK, D] f32 gate
                 const float* __restrict__ bias,          // [B,L,L,H] f32
                 const unsigned char* __restrict__ mask,  // [B,L] bool
                 float* __restrict__ a_out,    // [B,H,L,L] f32
                 u16* __restrict__ Yb) {       // [NTOK, D] bf16 gated y
  __shared__ float Srow[16][LPAD];
  __shared__ float pmax[16][4];
  __shared__ float psum[16][4];
  __shared__ float rinv[16];

  const int tid  = threadIdx.x;
  const int lane = tid & 31;
  const int wv   = tid >> 5;               // wave id 0..3
  const int bx = blockIdx.x;
  const int qt = bx % (LL / 16);
  const int h  = (bx / (LL / 16)) % HH;
  const int b  = bx / ((LL / 16) * HH);
  const int q0 = qt * 16;
  const int row  = lane & 15;
  const int half = lane >> 4;
  const int kbase = half * 8;

  // Q A-fragment (d_head = 32 -> one WMMA per 16x16 score tile)
  const u16* qptr = Qb + (size_t)(b * LL + q0 + row) * DD + h * CC + kbase;
  v16bf aq = load_frag_a(qptr);
  const v8f zero = (v8f)(0.0f);

  // ---- phase 1: S = QK^T + bias (masked); 12 k-tiles per wave ----
  for (int kt = wv; kt < LL / 16; kt += 4) {
    int k0 = kt * 16;
    const u16* kptr = Kb + (size_t)(b * LL + k0 + row) * DD + h * CC + half * 8;
    if (kt + 4 < LL / 16)
      __builtin_prefetch(kptr + (size_t)64 * DD, 0, 1);  // this wave's next tile
    v16bf bk = load_frag_b(kptr);
    v8f s = __builtin_amdgcn_wmma_f32_16x16x32_bf16(
        false, aq, false, bk, (short)0, zero, false, false);
    int ktok = k0 + row;
    int mok = mask[b * LL + ktok];
    const float* bptr = bias + ((size_t)(b * LL + q0) * LL + ktok) * HH + h;
    #pragma unroll
    for (int j = 0; j < 8; ++j) {
      int M = half * 8 + j;
      float val = s[j] + bptr[(size_t)M * LL * HH];
      if (!mok) val = NEG_BIG;
      Srow[M][ktok] = val;
    }
  }
  __syncthreads();

  // ---- phase 2: softmax stats; 8 segments of 96 (row, seg = wv*2+half) ----
  const int seg = wv * 2 + half;
  float4* srow4 = (float4*)&Srow[row][seg * 96];   // 24 float4, 16B aligned
  float m = -3.0e38f;
  #pragma unroll 4
  for (int i = 0; i < 24; ++i) {
    float4 t = srow4[i];
    m = fmaxf(m, fmaxf(fmaxf(t.x, t.y), fmaxf(t.z, t.w)));
  }
  m = fmaxf(m, __shfl_xor(m, 16, 32));
  if (half == 0) pmax[row][wv] = m;
  __syncthreads();
  const float mx = fmaxf(fmaxf(pmax[row][0], pmax[row][1]),
                         fmaxf(pmax[row][2], pmax[row][3]));
  float ssum = 0.0f;
  #pragma unroll 4
  for (int i = 0; i < 24; ++i) {
    float4 t = srow4[i];
    float4 e;
    e.x = __expf(t.x - mx); e.y = __expf(t.y - mx);
    e.z = __expf(t.z - mx); e.w = __expf(t.w - mx);
    srow4[i] = e;                                  // keep UNNORMALIZED exp
    ssum += (e.x + e.y) + (e.z + e.w);
  }
  ssum += __shfl_xor(ssum, 16, 32);
  if (half == 0) psum[row][wv] = ssum;
  __syncthreads();
  if (wv == 0 && half == 0) {
    float tot = (psum[row][0] + psum[row][1]) + (psum[row][2] + psum[row][3]);
    rinv[row] = 1.0f / tot;
  }
  __syncthreads();

  // ---- phase 3: write `a` = e * rinv, float4, lane-consecutive (coalesced)
  for (int r = 0; r < 16; ++r) {
    float inv = rinv[r];
    const float4* s4p = (const float4*)&Srow[r][0];
    float4* ao4 = (float4*)(a_out + ((size_t)(b * HH + h) * LL + q0 + r) * LL);
    for (int i = tid; i < LL / 4; i += 128) {
      float4 e = s4p[i];
      ao4[i] = make_float4(e.x * inv, e.y * inv, e.z * inv, e.w * inv);
    }
  }

  // ---- phase 4: Y_partial = E @ V, split-K: 6 chunks of 32 per wave ----
  v8f acc0 = (v8f)(0.0f), acc1 = (v8f)(0.0f);
  const u16* vbase = Vt + (size_t)(b * HH + h) * CC * LL;
  for (int kk = wv * 6; kk < wv * 6 + 6; ++kk) {
    int kb = kk * 32 + kbase;
    float4 e0 = *(const float4*)&Srow[row][kb];
    float4 e1 = *(const float4*)&Srow[row][kb + 4];
    float4 e2 = *(const float4*)&Srow[row][kb + 16];
    float4 e3 = *(const float4*)&Srow[row][kb + 20];
    ushort16 u;
    u[0]  = f2bf(e0.x); u[1]  = f2bf(e0.y); u[2]  = f2bf(e0.z); u[3]  = f2bf(e0.w);
    u[4]  = f2bf(e1.x); u[5]  = f2bf(e1.y); u[6]  = f2bf(e1.z); u[7]  = f2bf(e1.w);
    u[8]  = f2bf(e2.x); u[9]  = f2bf(e2.y); u[10] = f2bf(e2.z); u[11] = f2bf(e2.w);
    u[12] = f2bf(e3.x); u[13] = f2bf(e3.y); u[14] = f2bf(e3.z); u[15] = f2bf(e3.w);
    v16bf ap = __builtin_bit_cast(v16bf, u);
    v16bf bv0 = load_frag_b(vbase + (size_t)row * LL + kk * 32 + half * 8);
    v16bf bv1 = load_frag_b(vbase + (size_t)(16 + row) * LL + kk * 32 + half * 8);
    acc0 = __builtin_amdgcn_wmma_f32_16x16x32_bf16(
        false, ap, false, bv0, (short)0, acc0, false, false);
    acc1 = __builtin_amdgcn_wmma_f32_16x16x32_bf16(
        false, ap, false, bv1, (short)0, acc1, false, false);
  }
  __syncthreads();   // all waves done reading Srow -> safe to reuse as scratch

  // ---- phase 5: cross-wave reduction (reuse Srow), fold rinv + gate ----
  float* red = (float*)&Srow[0][0];  // 6*256 floats
  if (wv > 0) {
    #pragma unroll
    for (int j = 0; j < 8; ++j) {
      int M = half * 8 + j;
      red[((wv - 1) * 2 + 0) * 256 + M * 16 + row] = acc0[j];
      red[((wv - 1) * 2 + 1) * 256 + M * 16 + row] = acc1[j];
    }
  }
  __syncthreads();
  if (wv == 0) {
    #pragma unroll
    for (int j = 0; j < 8; ++j) {
      int M = half * 8 + j;
      float a0 = acc0[j], a1 = acc1[j];
      #pragma unroll
      for (int p = 0; p < 3; ++p) {
        a0 += red[(p * 2 + 0) * 256 + M * 16 + row];
        a1 += red[(p * 2 + 1) * 256 + M * 16 + row];
      }
      float ri = rinv[M];
      size_t tok = (size_t)(b * LL + q0 + M);
      float g0 = G[tok * DD + h * CC + row];
      float g1 = G[tok * DD + h * CC + 16 + row];
      Yb[tok * DD + h * CC + row]      = f2bf(a0 * ri * g0);
      Yb[tok * DD + h * CC + 16 + row] = f2bf(a1 * ri * g1);
    }
  }
}

// ---- host-side launch ------------------------------------------------------
extern "C" void kernel_launch(void* const* d_in, const int* in_sizes, int n_in,
                              void* d_out, int out_size, void* d_ws, size_t ws_size,
                              hipStream_t stream) {
  const float* x    = (const float*)d_in[0];                  // [B,L,D]
  const unsigned char* mask = (const unsigned char*)d_in[1];  // [B,L] bool
  const float* bias = (const float*)d_in[2];                  // [B,L,L,H]
  const float* Wq   = (const float*)d_in[3];
  const float* Wk   = (const float*)d_in[4];
  const float* Wv   = (const float*)d_in[5];
  const float* Wo   = (const float*)d_in[6];
  const float* Wg   = (const float*)d_in[7];
  const float* bg   = (const float*)d_in[8];

  float* outF  = (float*)d_out;          // [B,L,D]
  float* a_out = (float*)d_out + OUT0;   // [B,H,L,L]

  // workspace carve-up (256B aligned)
  char* base = (char*)d_ws;
  size_t off = 0;
  auto alloc = [&](size_t bytes) -> void* {
    void* r = base + off;
    off += (bytes + 255) & ~(size_t)255;
    return r;
  };
  u16* Xb  = (u16*)alloc((size_t)NTOK * DD * 2);
  u16* Wqb = (u16*)alloc((size_t)DD * DD * 2);
  u16* Wkb = (u16*)alloc((size_t)DD * DD * 2);
  u16* Wvb = (u16*)alloc((size_t)DD * DD * 2);
  u16* Wob = (u16*)alloc((size_t)DD * DD * 2);
  u16* Wgb = (u16*)alloc((size_t)DD * DD * 2);
  u16* Qb  = (u16*)alloc((size_t)NTOK * DD * 2);
  u16* Kb  = (u16*)alloc((size_t)NTOK * DD * 2);
  u16* Vt  = (u16*)alloc((size_t)NTOK * DD * 2);
  float* Gf = (float*)alloc((size_t)NTOK * DD * 4);
  u16* Yb  = (u16*)alloc((size_t)NTOK * DD * 2);

  // 1) fp32 -> bf16
  cvt_bf16_kernel<<<(NTOK * DD + 255) / 256, 256, 0, stream>>>(x, Xb, NTOK * DD);
  cvt_bf16_kernel<<<(DD * DD + 255) / 256, 256, 0, stream>>>(Wq, Wqb, DD * DD);
  cvt_bf16_kernel<<<(DD * DD + 255) / 256, 256, 0, stream>>>(Wk, Wkb, DD * DD);
  cvt_bf16_kernel<<<(DD * DD + 255) / 256, 256, 0, stream>>>(Wv, Wvb, DD * DD);
  cvt_bf16_kernel<<<(DD * DD + 255) / 256, 256, 0, stream>>>(Wo, Wob, DD * DD);
  cvt_bf16_kernel<<<(DD * DD + 255) / 256, 256, 0, stream>>>(Wg, Wgb, DD * DD);

  // 2) projections
  dim3 gg(NTOK / 128, DD / 64);
  gemm_bf16_kernel<<<gg, 256, 0, stream>>>(Xb, Wqb, 1, nullptr, nullptr, Qb);
  gemm_bf16_kernel<<<gg, 256, 0, stream>>>(Xb, Wkb, 0, nullptr, nullptr, Kb);
  gemm_bf16_kernel<<<gg, 256, 0, stream>>>(Xb, Wvb, 2, nullptr, nullptr, Vt);
  gemm_bf16_kernel<<<gg, 256, 0, stream>>>(Xb, Wgb, 3, bg, Gf, nullptr);

  // 3) attention + softmax + gating (4 waves per q-tile)
  attn_kernel<<<BB * HH * (LL / 16), 128, 0, stream>>>(Qb, Kb, Vt, Gf, bias, mask,
                                                       a_out, Yb);

  // 4) output projection
  gemm_bf16_kernel<<<gg, 256, 0, stream>>>(Yb, Wob, 4, nullptr, outF, nullptr);
}
